// ImgCrossAttn_15367392985116
// MI455X (gfx1250) — compile-verified, matrix operand order checked by reading, not run on hardware
//
#include <hip/hip_runtime.h>
#include <hip/hip_bf16.h>
#include <math.h>

// ---------------------------------------------------------------------------
// MI455X (gfx1250) implementation.
//
// Roofline: conv = 7.9e10 FLOPs dominates; K/V projections are algebraically
// collapsed (scores = S.(Wk^T q) + q.bk ; ctx = Wv.(attn@S) + bv) so the
// 5.7e10-FLOP per-key GEMMs become two 16000x128x128 GEMMs. All GEMMs run on
// v_wmma_f32_16x16x32_f16 (f16 in, f32 acc). img_h stored f16 channel-last
// (34MB, L2-resident at 192MB) so bilinear taps are contiguous v4h loads.
// Conv weights are staged per-block into LDS by the Tensor Data Mover
// (tensor_load_to_lds + s_wait_tensorcnt), shared by all 8 waves.
// ---------------------------------------------------------------------------

typedef __attribute__((ext_vector_type(16))) _Float16     v16h;
typedef __attribute__((ext_vector_type(8)))  float        v8f;
typedef __attribute__((ext_vector_type(4)))  _Float16     v4h;
typedef __attribute__((ext_vector_type(4)))  unsigned int v4u;
typedef __attribute__((ext_vector_type(8)))  int          v8i;
typedef __attribute__((ext_vector_type(4)))  int          v4i;

#define N_VOX   16000
#define PTS_C   64
#define IMG_C   256
#define HID     128
#define NCAM    6
#define IMGH    112
#define IMGW    200
#define PIX     (IMGH*IMGW)      // 22400
#define MAX_PTS 9
#define KEYS    (MAX_PTS*NCAM)   // 54
#define KCONV   (IMG_C*9)        // 2304
#define EPSV    1e-5f
#define NEGV    -1e30f

// 16-bit A/B operand K-index for lane/half per ISA 7.12.2 (16-bit A 16x32):
// lanes 0-15: halves 0..7 -> K0..7, halves 8..15 -> K16..23
// lanes 16-31: +8 on the base.
__device__ __forceinline__ int kmap(int h, int lane) {
  int base = (lane >> 4) << 3;
  return base + (h < 8 ? h : h + 8);
}

__device__ __forceinline__ v8f wmma16(v16h a, v16h b, v8f c) {
  return __builtin_amdgcn_wmma_f32_16x16x32_f16(false, a, false, b, (short)0, c,
                                                false, false);
}

// TDM 2D tile load: global (row-major, 16-bit elements) -> LDS, per D# spec
// (CDNA5 ISA). Dimensions in elements.
// clang-23 toolchain: 6-arg builtin (v4u, v8i, v4i, v4i, v8i, i32 cpol).
__device__ __forceinline__ void tdm_load_2d_f16(const void* gsrc,
                                                unsigned lds_off,
                                                unsigned tensor_d0,
                                                unsigned tensor_d1,
                                                unsigned tile_d0,
                                                unsigned tile_d1,
                                                unsigned stride_d0) {
  unsigned long long ga = (unsigned long long)(uintptr_t)gsrc;
  v4u g0;
  g0[0] = 1u;                                        // count=1 (valid user D#)
  g0[1] = lds_off;                                   // lds_addr (bytes)
  g0[2] = (unsigned)(ga & 0xffffffffu);              // global_addr[31:0]
  g0[3] = (unsigned)((ga >> 32) & 0x01ffffffu)       // global_addr[56:32]
        | 0x80000000u;                               // type=2 ("image")
  v8i g1;
  g1[0] = 0x00010000;                                // data_size=1 -> 2 bytes
  g1[1] = (int)((tensor_d0 & 0xffffu) << 16);        // tensor_dim0[15:0]
  g1[2] = (int)(((tensor_d0 >> 16) & 0xffffu) |      // tensor_dim0[31:16]
                ((tensor_d1 & 0xffffu) << 16));      // tensor_dim1[15:0]
  g1[3] = (int)(((tensor_d1 >> 16) & 0xffffu) |      // tensor_dim1[31:16]
                ((tile_d0 & 0xffffu) << 16));        // tile_dim0
  g1[4] = (int)(tile_d1 & 0xffffu);                  // tile_dim1 (tile_dim2=0)
  g1[5] = (int)stride_d0;                            // tensor_dim0_stride lo
  g1[6] = 0;                                         // stride hi / dim1_stride
  g1[7] = 0;
  v4i z4 = {};
  v8i z8 = {};
  __builtin_amdgcn_tensor_load_to_lds(g0, g1, z4, z4, z8, 0);
}

// ---------------------------------------------------------------------------
// K0: convert weights to f16 scratch copies.
// ---------------------------------------------------------------------------
__global__ void prep_weights(const float* __restrict__ W_img,
                             const float* __restrict__ W_pts,
                             const float* __restrict__ in_proj_w,
                             const float* __restrict__ out_w,
                             _Float16* wimg, _Float16* wpts, _Float16* wq,
                             _Float16* wk, _Float16* wv, _Float16* wout) {
  int i = blockIdx.x * blockDim.x + threadIdx.x;
  int stride = gridDim.x * blockDim.x;
  for (int t = i; t < HID * KCONV; t += stride) wimg[t] = (_Float16)W_img[t];
  for (int t = i; t < PTS_C * HID; t += stride) wpts[t] = (_Float16)W_pts[t];
  for (int t = i; t < HID * HID; t += stride) {
    wq[t]   = (_Float16)in_proj_w[t];
    wk[t]   = (_Float16)in_proj_w[HID * HID + t];
    wv[t]   = (_Float16)in_proj_w[2 * HID * HID + t];
    wout[t] = (_Float16)out_w[t];
  }
}

// ---------------------------------------------------------------------------
// K1: 3x3 conv as implicit GEMM. One wave -> 16 pixels x 128 out-channels.
// A = im2col(img_feats) f16, B = W_img f16 staged in LDS by TDM, C f32.
// Output channel-last f16. Grid is exactly 1050 blocks x 8 waves.
// ---------------------------------------------------------------------------
__global__ __launch_bounds__(256) void conv_wmma(
    const float* __restrict__ img_feats, const _Float16* __restrict__ wimg,
    const float* __restrict__ b_img, _Float16* __restrict__ img_h) {
  __shared__ _Float16 btile[32 * HID];           // [oc][k_local] 8 KB
  const int lane = threadIdx.x & 31;
  const int wave = threadIdx.x >> 5;
  const int tile = blockIdx.x * 8 + wave;        // NCAM*PIX/16 = 8400 tiles
  const int cam   = tile / (PIX / 16);
  const int ptile = tile % (PIX / 16);
  const int col   = lane & 15;                   // A-row / B-col / D-col lane idx
  const int p     = ptile * 16 + col;
  const int py    = p / IMGW, px = p % IMGW;
  const float* imgc = img_feats + (size_t)cam * IMG_C * PIX;
  const unsigned lds_off = (unsigned)(uintptr_t)&btile[0];

  v8f acc[8] = {};
  for (int kk = 0; kk < KCONV; kk += 32) {
    __syncthreads();                             // previous slab fully consumed
    if (wave == 0) {
      // Stage 32(K) x 128(oc) W_img slab into LDS via Tensor Data Mover.
      tdm_load_2d_f16(wimg + kk, lds_off, /*tensor_d0=*/KCONV,
                      /*tensor_d1=*/HID, /*tile_d0=*/32, /*tile_d1=*/HID,
                      /*stride_d0=*/KCONV);
      __builtin_amdgcn_s_wait_tensorcnt(0);
    }
    __syncthreads();                             // slab visible to all waves

    // prefetch next K-slab of im2col source (emits global_prefetch_b8)
    if (kk + 32 < KCONV) {
      int icn = (kk + 32) / 9;
      __builtin_prefetch(imgc + (size_t)icn * PIX + py * IMGW + px, 0, 1);
    }
    v16h a;
#pragma unroll
    for (int h = 0; h < 16; ++h) {
      int k  = kk + kmap(h, lane);
      int ic = k / 9, r9 = k - ic * 9;
      int kh = r9 / 3, kw = r9 - kh * 3;
      int yy = py + kh - 1, xx = px + kw - 1;
      float v = 0.f;
      if (yy >= 0 && yy < IMGH && xx >= 0 && xx < IMGW)
        v = imgc[(size_t)ic * PIX + yy * IMGW + xx];
      a[h] = (_Float16)v;
    }
#pragma unroll
    for (int nt = 0; nt < 8; ++nt) {
      const int n = nt * 16 + col;
      v16h b;
#pragma unroll
      for (int h = 0; h < 16; ++h) b[h] = btile[n * 32 + kmap(h, lane)];
      acc[nt] = wmma16(a, b, acc[nt]);
    }
  }
  const int hi = lane >> 4;
#pragma unroll
  for (int nt = 0; nt < 8; ++nt) {
    int oc = nt * 16 + col;
    float bi = b_img[oc];
#pragma unroll
    for (int r = 0; r < 8; ++r) {
      int pp = ptile * 16 + r + 8 * hi;
      img_h[((size_t)cam * PIX + pp) * HID + oc] = (_Float16)(acc[nt][r] + bi);
    }
  }
}

// ---------------------------------------------------------------------------
// K2: per 16-voxel tile (one wave), three chained WMMA GEMMs through LDS:
//   vf = voxel_feats @ W_pts + b_pts           (store f32, stage f16 in LDS)
//   q  = (vf @ Wq^T + bq) * scale              (stage f16 in LDS)
//   T  = q @ Wk   (i.e. Wk^T q per voxel)      (store f16)
// plus bkq[v] = bk . q[v] via wave shuffle reduction.
// ---------------------------------------------------------------------------
__global__ __launch_bounds__(256) void vf_q_t_wmma(
    const float* __restrict__ voxel_feats, const _Float16* __restrict__ wpts,
    const float* __restrict__ b_pts, const _Float16* __restrict__ wq,
    const _Float16* __restrict__ wk, const float* __restrict__ in_proj_b,
    float* __restrict__ vf_out, _Float16* __restrict__ T_out,
    float* __restrict__ bkq_out) {
  __shared__ _Float16 lds[8][16][HID + 8];
  const int lane = threadIdx.x & 31;
  const int wave = threadIdx.x >> 5;
  const int tile = blockIdx.x * 8 + wave;        // 1000 tiles over 125 blocks
  const int vbase = tile * 16;
  const int col = lane & 15;
  const int hi = lane >> 4;
  const float scale = 0.08838834764831845f;      // 1/sqrt(128)

  // ---- stage 1: vf ----
  v8f acc[8] = {};
  for (int kk = 0; kk < PTS_C; kk += 32) {
    v16h a;
#pragma unroll
    for (int h = 0; h < 16; ++h) {
      int k = kk + kmap(h, lane);
      a[h] = (_Float16)voxel_feats[(size_t)(vbase + col) * PTS_C + k];
    }
#pragma unroll
    for (int nt = 0; nt < 8; ++nt) {
      v16h b;
      int n = nt * 16 + col;
#pragma unroll
      for (int h = 0; h < 16; ++h) b[h] = wpts[(kk + kmap(h, lane)) * HID + n];
      acc[nt] = wmma16(a, b, acc[nt]);
    }
  }
#pragma unroll
  for (int nt = 0; nt < 8; ++nt) {
    int oc = nt * 16 + col;
    float bi = b_pts[oc];
#pragma unroll
    for (int r = 0; r < 8; ++r) {
      int mm = r + 8 * hi;
      float v = acc[nt][r] + bi;
      vf_out[(size_t)(vbase + mm) * HID + oc] = v;
      lds[wave][mm][oc] = (_Float16)v;
    }
  }
  __syncthreads();

  // ---- stage 2: q = vf @ Wq^T ----
  v8f qa[8] = {};
  for (int kk = 0; kk < HID; kk += 32) {
    v16h a;
#pragma unroll
    for (int h = 0; h < 16; ++h) a[h] = lds[wave][col][kk + kmap(h, lane)];
#pragma unroll
    for (int nt = 0; nt < 8; ++nt) {
      v16h b;
      int n = nt * 16 + col;
#pragma unroll
      for (int h = 0; h < 16; ++h) b[h] = wq[n * HID + kk + kmap(h, lane)];
      qa[nt] = wmma16(a, b, qa[nt]);
    }
  }
  __syncthreads();
#pragma unroll
  for (int nt = 0; nt < 8; ++nt) {
    int oc = nt * 16 + col;
    float bq = in_proj_b[oc];
#pragma unroll
    for (int r = 0; r < 8; ++r) {
      int mm = r + 8 * hi;
      lds[wave][mm][oc] = (_Float16)((qa[nt][r] + bq) * scale);
    }
  }
  __syncthreads();

  // ---- bkq[v] = bk . q[v] ----
  for (int mm = 0; mm < 16; ++mm) {
    float part = 0.f;
#pragma unroll
    for (int t = 0; t < 4; ++t) {
      int c = lane + 32 * t;
      part += in_proj_b[HID + c] * (float)lds[wave][mm][c];
    }
    part += __shfl_xor(part, 16);
    part += __shfl_xor(part, 8);
    part += __shfl_xor(part, 4);
    part += __shfl_xor(part, 2);
    part += __shfl_xor(part, 1);
    if (lane == 0) bkq_out[vbase + mm] = part;
  }

  // ---- stage 3: T = q @ Wk (reduction over q-rows of Wk) ----
  v8f ta[8] = {};
  for (int kk = 0; kk < HID; kk += 32) {
    v16h a;
#pragma unroll
    for (int h = 0; h < 16; ++h) a[h] = lds[wave][col][kk + kmap(h, lane)];
#pragma unroll
    for (int nt = 0; nt < 8; ++nt) {
      v16h b;
      int n = nt * 16 + col;
#pragma unroll
      for (int h = 0; h < 16; ++h) b[h] = wk[(kk + kmap(h, lane)) * HID + n];
      ta[nt] = wmma16(a, b, ta[nt]);
    }
  }
#pragma unroll
  for (int nt = 0; nt < 8; ++nt) {
    int oc = nt * 16 + col;
#pragma unroll
    for (int r = 0; r < 8; ++r) {
      int mm = r + 8 * hi;
      T_out[(size_t)(vbase + mm) * HID + oc] = (_Float16)ta[nt][r];
    }
  }
}

// ---------------------------------------------------------------------------
// K3: one wave per voxel. Project 54 points, bilinear-sample img_h into LDS,
// scores = S.t + bkq, masked softmax, U = sum_j attn_j * S_j.
// ---------------------------------------------------------------------------
__global__ __launch_bounds__(64) void sample_attn(
    const int* __restrict__ voxel_inds, const float* __restrict__ lidar2img,
    const _Float16* __restrict__ img_h, const _Float16* __restrict__ T_in,
    const float* __restrict__ bkq_in, _Float16* __restrict__ u_out,
    int* __restrict__ valid_out) {
  __shared__ _Float16 S[2][KEYS][HID + 4];
  __shared__ float kix[2][KEYS], kiy[2][KEYS], sc[2][KEYS];
  __shared__ int km[2][KEYS];
  const int lane = threadIdx.x & 31;
  const int w = threadIdx.x >> 5;
  const int v = blockIdx.x * 2 + w;              // exactly 8000 blocks

  const float cz = (float)voxel_inds[v * 4 + 1];
  const float cy = (float)voxel_inds[v * 4 + 2];
  const float cx = (float)voxel_inds[v * 4 + 3];

  // corner offsets (z,y,x), reference ordering [0,1,3,2,4,5,7,6]
  const float OZ[8] = {-.25f,-.25f,-.25f,-.25f, .25f, .25f, .25f, .25f};
  const float OY[8] = {-.25f,-.25f, .25f, .25f,-.25f,-.25f, .25f, .25f};
  const float OX[8] = {-.25f, .25f, .25f,-.25f,-.25f, .25f, .25f,-.25f};

  for (int j = lane; j < KEYS; j += 32) {
    int pt = j / NCAM, cam = j % NCAM;
    float oz = 0.f, oy = 0.f, ox = 0.f;
    if (pt > 0) { oz = OZ[pt - 1]; oy = OY[pt - 1]; ox = OX[pt - 1]; }
    float z = (cz + oz + 0.5f) * 1.6f - 5.0f;    // VOXEL_SIZE.z * RATIO
    float y = (cy + oy + 0.5f) * 0.6f - 54.0f;
    float x = (cx + ox + 0.5f) * 0.6f - 54.0f;
    const float* M = lidar2img + cam * 16;
    float c0 = M[0] * x + M[1] * y + M[2]  * z + M[3];
    float c1 = M[4] * x + M[5] * y + M[6]  * z + M[7];
    float c2 = M[8] * x + M[9] * y + M[10] * z + M[11];
    int ok = c2 > EPSV;
    float d = fmaxf(c2, EPSV);
    float gx = (c0 / d) * (1.0f / 1600.0f);
    float gy = (c1 / d) * (1.0f / 900.0f);
    gx = (gx - 0.5f) * 2.f;
    gy = (gy - 0.5f) * 2.f;
    ok = ok && (gx > -1.f) && (gx < 1.f) && (gy > -1.f) && (gy < 1.f);
    float ix = ((gx + 1.f) * IMGW - 1.f) * 0.5f;
    float iy = ((gy + 1.f) * IMGH - 1.f) * 0.5f;
    ix = fminf(fmaxf(ix, -4.0f), 4.0f + IMGW);   // tame wild OOB coords
    iy = fminf(fmaxf(iy, -4.0f), 4.0f + IMGH);
    kix[w][j] = ix; kiy[w][j] = iy; km[w][j] = ok;
  }
  __syncthreads();

  // bilinear sample: lane owns 4 contiguous channels
  const int cbase = lane * 4;
  for (int j = 0; j < KEYS; ++j) {
    float ix = kix[w][j], iy = kiy[w][j];
    float x0f = floorf(ix), y0f = floorf(iy);
    int x0 = (int)x0f, y0 = (int)y0f;
    float wx1 = ix - x0f, wx0 = 1.f - wx1;
    float wy1 = iy - y0f, wy0 = 1.f - wy1;
    const _Float16* base = img_h + (size_t)(j % NCAM) * PIX * HID;
    float a0 = 0.f, a1 = 0.f, a2 = 0.f, a3 = 0.f;
#pragma unroll
    for (int t = 0; t < 4; ++t) {
      int xi = x0 + (t & 1), yi = y0 + (t >> 1);
      float wgt = ((t & 1) ? wx1 : wx0) * ((t >> 1) ? wy1 : wy0);
      if (xi >= 0 && xi < IMGW && yi >= 0 && yi < IMGH) {
        const v4h tp = *(const v4h*)(base + ((size_t)yi * IMGW + xi) * HID + cbase);
        a0 += wgt * (float)tp[0]; a1 += wgt * (float)tp[1];
        a2 += wgt * (float)tp[2]; a3 += wgt * (float)tp[3];
      }
    }
    v4h sv; sv[0] = (_Float16)a0; sv[1] = (_Float16)a1;
    sv[2] = (_Float16)a2; sv[3] = (_Float16)a3;
    *(v4h*)&S[w][j][cbase] = sv;
  }
  __syncthreads();

  // scores = S . t + bkq
  float t0 = (float)T_in[(size_t)v * HID + cbase + 0];
  float t1 = (float)T_in[(size_t)v * HID + cbase + 1];
  float t2 = (float)T_in[(size_t)v * HID + cbase + 2];
  float t3 = (float)T_in[(size_t)v * HID + cbase + 3];
  float bq = bkq_in[v];
  for (int j = 0; j < KEYS; ++j) {
    float part = t0 * (float)S[w][j][cbase + 0] + t1 * (float)S[w][j][cbase + 1] +
                 t2 * (float)S[w][j][cbase + 2] + t3 * (float)S[w][j][cbase + 3];
    part += __shfl_xor(part, 16);
    part += __shfl_xor(part, 8);
    part += __shfl_xor(part, 4);
    part += __shfl_xor(part, 2);
    part += __shfl_xor(part, 1);
    if (lane == 0) sc[w][j] = part + bq;
  }
  __syncthreads();

  // masked softmax over 54 keys (lanes cover j=lane and j=lane+32)
  int j2 = lane + 32;
  float s1 = (lane < KEYS && km[w][lane]) ? sc[w][lane] : NEGV;
  float s2 = (j2 < KEYS && km[w][j2]) ? sc[w][j2] : NEGV;
  float mx = fmaxf(s1, s2);
  mx = fmaxf(mx, __shfl_xor(mx, 16));
  mx = fmaxf(mx, __shfl_xor(mx, 8));
  mx = fmaxf(mx, __shfl_xor(mx, 4));
  mx = fmaxf(mx, __shfl_xor(mx, 2));
  mx = fmaxf(mx, __shfl_xor(mx, 1));
  float e1 = (lane < KEYS) ? __expf(s1 - mx) : 0.f;
  float e2 = (j2 < KEYS) ? __expf(s2 - mx) : 0.f;
  float sum = e1 + e2;
  sum += __shfl_xor(sum, 16);
  sum += __shfl_xor(sum, 8);
  sum += __shfl_xor(sum, 4);
  sum += __shfl_xor(sum, 2);
  sum += __shfl_xor(sum, 1);
  float inv = 1.f / sum;
  int anyv = (lane < KEYS && km[w][lane]) || (j2 < KEYS && km[w][j2]);
  anyv = __any(anyv);
  __syncthreads();
  if (lane < KEYS) sc[w][lane] = e1 * inv;
  if (j2 < KEYS) sc[w][j2] = e2 * inv;
  if (lane == 0) valid_out[v] = anyv;
  __syncthreads();

  // U = sum_j attn_j * S_j
  float u0 = 0.f, u1 = 0.f, u2 = 0.f, u3 = 0.f;
  for (int j = 0; j < KEYS; ++j) {
    float aj = sc[w][j];
    u0 += aj * (float)S[w][j][cbase + 0];
    u1 += aj * (float)S[w][j][cbase + 1];
    u2 += aj * (float)S[w][j][cbase + 2];
    u3 += aj * (float)S[w][j][cbase + 3];
  }
  v4h uv; uv[0] = (_Float16)u0; uv[1] = (_Float16)u1;
  uv[2] = (_Float16)u2; uv[3] = (_Float16)u3;
  *(v4h*)(u_out + (size_t)v * HID + cbase) = uv;
}

// ---------------------------------------------------------------------------
// K4: ctx = Wv u + bv ; out = ctx @ out_w^T + out_b ; select valid ? out : vf
// ---------------------------------------------------------------------------
__global__ __launch_bounds__(256) void out_wmma(
    const _Float16* __restrict__ u_in, const _Float16* __restrict__ wv,
    const float* __restrict__ in_proj_b, const _Float16* __restrict__ wout,
    const float* __restrict__ out_b, const float* __restrict__ vf_in,
    const int* __restrict__ valid_in, float* __restrict__ out) {
  __shared__ _Float16 lds[8][16][HID + 8];
  const int lane = threadIdx.x & 31;
  const int wave = threadIdx.x >> 5;
  const int tile = blockIdx.x * 8 + wave;
  const int vbase = tile * 16;
  const int col = lane & 15;
  const int hi = lane >> 4;

  // ---- stage 1: ctx = U @ Wv^T (per-voxel Wv.u) + bv ----
  v8f ca[8] = {};
  for (int kk = 0; kk < HID; kk += 32) {
    v16h a;
#pragma unroll
    for (int h = 0; h < 16; ++h)
      a[h] = u_in[(size_t)(vbase + col) * HID + kk + kmap(h, lane)];
#pragma unroll
    for (int nt = 0; nt < 8; ++nt) {
      v16h b;
      int n = nt * 16 + col;
#pragma unroll
      for (int h = 0; h < 16; ++h) b[h] = wv[n * HID + kk + kmap(h, lane)];
      ca[nt] = wmma16(a, b, ca[nt]);
    }
  }
#pragma unroll
  for (int nt = 0; nt < 8; ++nt) {
    int oc = nt * 16 + col;
    float bv = in_proj_b[2 * HID + oc];
#pragma unroll
    for (int r = 0; r < 8; ++r) {
      int mm = r + 8 * hi;
      lds[wave][mm][oc] = (_Float16)(ca[nt][r] + bv);
    }
  }
  __syncthreads();

  // ---- stage 2: out = ctx @ out_w^T + out_b, fallback to vf ----
  v8f oa[8] = {};
  for (int kk = 0; kk < HID; kk += 32) {
    v16h a;
#pragma unroll
    for (int h = 0; h < 16; ++h) a[h] = lds[wave][col][kk + kmap(h, lane)];
#pragma unroll
    for (int nt = 0; nt < 8; ++nt) {
      v16h b;
      int n = nt * 16 + col;
#pragma unroll
      for (int h = 0; h < 16; ++h) b[h] = wout[n * HID + kk + kmap(h, lane)];
      oa[nt] = wmma16(a, b, oa[nt]);
    }
  }
#pragma unroll
  for (int nt = 0; nt < 8; ++nt) {
    int oc = nt * 16 + col;
    float bo = out_b[oc];
#pragma unroll
    for (int r = 0; r < 8; ++r) {
      int mm = r + 8 * hi;
      int vv = vbase + mm;
      float val = oa[nt][r] + bo;
      float fb = vf_in[(size_t)vv * HID + oc];
      out[(size_t)vv * HID + oc] = valid_in[vv] ? val : fb;
    }
  }
}

// ---------------------------------------------------------------------------
extern "C" void kernel_launch(void* const* d_in, const int* in_sizes, int n_in,
                              void* d_out, int out_size, void* d_ws,
                              size_t ws_size, hipStream_t stream) {
  const float* voxel_feats = (const float*)d_in[0];
  const int*   voxel_inds  = (const int*)d_in[1];
  const float* img_feats   = (const float*)d_in[2];
  const float* lidar2img   = (const float*)d_in[3];
  const float* W_pts       = (const float*)d_in[4];
  const float* b_pts       = (const float*)d_in[5];
  const float* W_img       = (const float*)d_in[6];
  const float* b_img       = (const float*)d_in[7];
  const float* in_proj_w   = (const float*)d_in[8];
  const float* in_proj_b   = (const float*)d_in[9];
  const float* out_w       = (const float*)d_in[10];
  const float* out_b       = (const float*)d_in[11];
  float* out = (float*)d_out;

  char* ws = (char*)d_ws;
  size_t off = 0;
  auto alloc = [&](size_t bytes) {
    size_t o = off;
    off += (bytes + 255) & ~(size_t)255;
    return o;
  };
  _Float16* wimg  = (_Float16*)(ws + alloc((size_t)HID * KCONV * 2));
  _Float16* wpts  = (_Float16*)(ws + alloc((size_t)PTS_C * HID * 2));
  _Float16* wq    = (_Float16*)(ws + alloc((size_t)HID * HID * 2));
  _Float16* wk    = (_Float16*)(ws + alloc((size_t)HID * HID * 2));
  _Float16* wv    = (_Float16*)(ws + alloc((size_t)HID * HID * 2));
  _Float16* wout  = (_Float16*)(ws + alloc((size_t)HID * HID * 2));
  _Float16* img_h = (_Float16*)(ws + alloc((size_t)NCAM * PIX * HID * 2));
  float*    vf    = (float*)(ws + alloc((size_t)N_VOX * HID * 4));
  _Float16* T     = (_Float16*)(ws + alloc((size_t)N_VOX * HID * 2));
  float*    bkq   = (float*)(ws + alloc((size_t)N_VOX * 4));
  _Float16* u     = (_Float16*)(ws + alloc((size_t)N_VOX * HID * 2));
  int*      valid = (int*)(ws + alloc((size_t)N_VOX * 4));
  (void)ws_size; (void)in_sizes; (void)n_in; (void)out_size;

  prep_weights<<<256, 256, 0, stream>>>(W_img, W_pts, in_proj_w, out_w, wimg,
                                        wpts, wq, wk, wv, wout);
  conv_wmma<<<(NCAM * (PIX / 16)) / 8, 256, 0, stream>>>(img_feats, wimg,
                                                         b_img, img_h);
  vf_q_t_wmma<<<(N_VOX / 16) / 8, 256, 0, stream>>>(
      voxel_feats, wpts, b_pts, wq, wk, in_proj_b, vf, T, bkq);
  sample_attn<<<N_VOX / 2, 64, 0, stream>>>(voxel_inds, lidar2img, img_h, T,
                                            bkq, u, valid);
  out_wmma<<<(N_VOX / 16) / 8, 256, 0, stream>>>(u, wv, in_proj_b, wout, out_b,
                                                 vf, valid, out);
}
